// DNAGraphNetwork_62783831933588
// MI455X (gfx1250) — compile-verified, matrix operand order checked by reading, not run on hardware
//
#include <hip/hip_runtime.h>
#include <stdint.h>

// ---------------------------------------------------------------------------
// GCN (3-layer, 128-ch) + mean-pool + MLP head for MI455X (gfx1250, wave32).
// GEMMs: v_wmma_f32_16x16x32_bf16, 2 M-tiles per wave (B-fragment reuse),
// self-loop term fused into the GEMM epilogue. Edge gather/scatter: coalesced
// float4 reads + global f32 atomics (node features 51MB << 192MB L2).
// ---------------------------------------------------------------------------

#define N_NODES   100000
#define N_EDGES   3200000
#define HID       128
#define NUM_GRAPHS 64
#define OUT_CLASSES 5
#define N_TILES32 (N_NODES / 32)   // 3125, exact

typedef __attribute__((ext_vector_type(16))) __bf16 v16bf;
typedef __attribute__((ext_vector_type(8)))  __bf16 v8bf;
typedef __attribute__((ext_vector_type(8)))  float  v8f;

__device__ __forceinline__ unsigned short f2bf(float f) {
    unsigned int u = __float_as_uint(f);
    unsigned int r = (u + 0x7FFFu + ((u >> 16) & 1u)) >> 16;  // RNE
    return (unsigned short)r;
}

// ---------------- degree / norm precompute ----------------

__global__ void k_init_deg(float* __restrict__ deg) {
    int n = blockIdx.x * blockDim.x + threadIdx.x;
    if (n < N_NODES) deg[n] = 1.0f;               // self loop
}

__global__ void k_deg_acc(const int* __restrict__ dst, float* __restrict__ deg) {
    int e = blockIdx.x * blockDim.x + threadIdx.x;
    if (e < N_EDGES) atomicAdd(&deg[dst[e]], 1.0f);
}

__global__ void k_norms(const float* __restrict__ deg,
                        float* __restrict__ dinv, float* __restrict__ invdeg) {
    int n = blockIdx.x * blockDim.x + threadIdx.x;
    if (n < N_NODES) {
        float d = deg[n];
        dinv[n]   = rsqrtf(d);
        invdeg[n] = 1.0f / d;
    }
}

__global__ void k_edge_norm(const int* __restrict__ src, const int* __restrict__ dst,
                            const float* __restrict__ dinv, float* __restrict__ en) {
    int e = blockIdx.x * blockDim.x + threadIdx.x;
    if (e < N_EDGES) en[e] = dinv[src[e]] * dinv[dst[e]];
}

// ---------------- bf16 conversion / weight packing ----------------

__global__ void k_to_bf16_v4(const float* __restrict__ src, unsigned short* __restrict__ dst,
                             int count4) {
    int i = blockIdx.x * blockDim.x + threadIdx.x;   // one float4 per thread
    if (i >= count4) return;
    float4 v = *(const float4*)(src + (size_t)i * 4);
    ushort4 o;
    o.x = f2bf(v.x); o.y = f2bf(v.y); o.z = f2bf(v.z); o.w = f2bf(v.w);
    *(ushort4*)(dst + (size_t)i * 4) = o;
}

// Pack W[128x128] (rows=K, cols=N) into WMMA B-fragment order: one contiguous
// 32B v16bf per lane per (kt,nt) tile (layout per ISA 7.12.2, 16-bit operands).
__global__ void k_pack_w(const float* __restrict__ W, unsigned short* __restrict__ wpack) {
    int t = blockIdx.x * blockDim.x + threadIdx.x;   // 0 .. 16383
    if (t >= HID * HID) return;
    int i    = t & 15;          // element within lane fragment
    int lane = (t >> 4) & 31;
    int nt   = (t >> 9) & 7;
    int kt   = t >> 12;
    int half = lane >> 4;       // 0: K base 0 / 1: K base 8
    int k = kt * 32 + ((i < 8) ? (half * 8 + i) : (16 + half * 8 + (i - 8)));
    int n = nt * 16 + (lane & 15);
    wpack[t] = f2bf(W[k * HID + n]);
}

// ---- WMMA GEMM, 32 rows/wave, fused self-loop epilogue ----
// hW  = hb @ W              (needed by edge gather)
// agg = (hb @ W) * deg^-1   (self-loop init of the scatter accumulator)

__global__ __launch_bounds__(128) void k_gemm_wmma(
    const unsigned short* __restrict__ hb,
    const unsigned short* __restrict__ wpack,
    const float* __restrict__ invdeg,
    float* __restrict__ hW, float* __restrict__ agg, int n_tiles32) {
    int wave = threadIdx.x >> 5;
    int lane = threadIdx.x & 31;
    int tile = blockIdx.x * 4 + wave;
    if (tile >= n_tiles32) return;

    int m    = lane & 15;
    int half = lane >> 4;
    const __bf16* hrow0 = (const __bf16*)hb + (size_t)(tile * 32 + m) * HID;
    const __bf16* hrow1 = hrow0 + (size_t)16 * HID;
    const v16bf*  wp    = (const v16bf*)wpack;

    v8f acc0[8], acc1[8];
#pragma unroll
    for (int i = 0; i < 8; ++i) { acc0[i] = (v8f)0.0f; acc1[i] = (v8f)0.0f; }

#pragma unroll
    for (int kt = 0; kt < 4; ++kt) {
        union { v16bf v; v8bf h[2]; } a0, a1;
        int klo = kt * 32 + half * 8;
        a0.h[0] = *(const v8bf*)(hrow0 + klo);
        a0.h[1] = *(const v8bf*)(hrow0 + klo + 16);
        a1.h[0] = *(const v8bf*)(hrow1 + klo);
        a1.h[1] = *(const v8bf*)(hrow1 + klo + 16);
#pragma unroll
        for (int nt = 0; nt < 8; ++nt) {
            v16bf b = wp[(kt * 8 + nt) * 32 + lane];
            acc0[nt] = __builtin_amdgcn_wmma_f32_16x16x32_bf16(
                false, a0.v, false, b, (short)0, acc0[nt], false, false);
            acc1[nt] = __builtin_amdgcn_wmma_f32_16x16x32_bf16(
                false, a1.v, false, b, (short)0, acc1[nt], false, false);
        }
    }

    // C/D layout: lanes 0-15 -> M=v, lanes 16-31 -> M=v+8; N = lane&15.
    int mbase0 = tile * 32 + half * 8;
    int mbase1 = mbase0 + 16;
    int col    = lane & 15;
    float id0[8], id1[8];
#pragma unroll
    for (int v = 0; v < 8; ++v) { id0[v] = invdeg[mbase0 + v]; id1[v] = invdeg[mbase1 + v]; }
#pragma unroll
    for (int nt = 0; nt < 8; ++nt) {
#pragma unroll
        for (int v = 0; v < 8; ++v) {
            size_t o0 = (size_t)(mbase0 + v) * HID + nt * 16 + col;
            size_t o1 = (size_t)(mbase1 + v) * HID + nt * 16 + col;
            float c0 = acc0[nt][v], c1 = acc1[nt][v];
            hW[o0]  = c0;          hW[o1]  = c1;
            agg[o0] = c0 * id0[v]; agg[o1] = c1 * id1[v];
        }
    }
}

// ---------------- aggregation ----------------

// one wave per edge; 32 lanes x float4 = full 128-ch row, coalesced 512B read
__global__ void k_edge_scatter(const float* __restrict__ hW, const float* __restrict__ en,
                               const int* __restrict__ src, const int* __restrict__ dst,
                               float* __restrict__ agg) {
    long long t = (long long)blockIdx.x * blockDim.x + threadIdx.x;
    int e = (int)(t >> 5);
    if (e >= N_EDGES) return;
    int c = ((int)t & 31) * 4;
    float w = en[e];
    int s = src[e], d = dst[e];
    float4 v = *(const float4*)(hW + (size_t)s * HID + c);
    float* ap = agg + (size_t)d * HID + c;
    atomicAdd(ap + 0, v.x * w);
    atomicAdd(ap + 1, v.y * w);
    atomicAdd(ap + 2, v.z * w);
    atomicAdd(ap + 3, v.w * w);
}

// h = relu(agg + b); also emit bf16 copy for the next layer's GEMM
__global__ void k_relu_bias(const float* __restrict__ agg, const float* __restrict__ bias,
                            float* __restrict__ h, unsigned short* __restrict__ hb,
                            int write_bf16) {
    int i = blockIdx.x * blockDim.x + threadIdx.x;   // one float4 per thread
    if (i >= N_NODES * (HID / 4)) return;
    int cb = (i * 4) & (HID - 1);
    float4 v = *(const float4*)(agg + (size_t)i * 4);
    v.x = fmaxf(v.x + bias[cb + 0], 0.0f);
    v.y = fmaxf(v.y + bias[cb + 1], 0.0f);
    v.z = fmaxf(v.z + bias[cb + 2], 0.0f);
    v.w = fmaxf(v.w + bias[cb + 3], 0.0f);
    *(float4*)(h + (size_t)i * 4) = v;
    if (write_bf16) {
        ushort4 o;
        o.x = f2bf(v.x); o.y = f2bf(v.y); o.z = f2bf(v.z); o.w = f2bf(v.w);
        *(ushort4*)(hb + (size_t)i * 4) = o;
    }
}

// ---------------- pooling + head ----------------

__global__ void k_zero(float* __restrict__ p, int count) {
    int i = blockIdx.x * blockDim.x + threadIdx.x;
    if (i < count) p[i] = 0.0f;
}

__global__ void k_pool_acc(const float* __restrict__ h, const int* __restrict__ batch,
                           float* __restrict__ sums) {
    long long t = (long long)blockIdx.x * blockDim.x + threadIdx.x;
    int n = (int)(t >> 5);
    if (n >= N_NODES) return;
    int c = ((int)t & 31) * 4;
    int g = batch[n];
    float4 v = *(const float4*)(h + (size_t)n * HID + c);
    float* sp = sums + (size_t)g * HID + c;
    atomicAdd(sp + 0, v.x);
    atomicAdd(sp + 1, v.y);
    atomicAdd(sp + 2, v.z);
    atomicAdd(sp + 3, v.w);
}

__global__ void k_cnt_acc(const int* __restrict__ batch, float* __restrict__ cnt) {
    int n = blockIdx.x * blockDim.x + threadIdx.x;
    if (n < N_NODES) atomicAdd(&cnt[batch[n]], 1.0f);
}

__global__ __launch_bounds__(128) void k_head(
    const float* __restrict__ sums, const float* __restrict__ cnt,
    const float* __restrict__ Wm1, const float* __restrict__ bm1,
    const float* __restrict__ Wm2, const float* __restrict__ bm2,
    float* __restrict__ out) {
    __shared__ float pooled[HID];
    __shared__ float t1[HID];
    int g = blockIdx.x, tid = threadIdx.x;
    pooled[tid] = sums[(size_t)g * HID + tid] / fmaxf(cnt[g], 1.0f);
    __syncthreads();
    float a = bm1[tid];
    for (int k = 0; k < HID; ++k) a += pooled[k] * Wm1[k * HID + tid];
    t1[tid] = fmaxf(a, 0.0f);
    __syncthreads();
    if (tid < OUT_CLASSES) {
        float o = bm2[tid];
        for (int k = 0; k < HID; ++k) o += t1[k] * Wm2[k * OUT_CLASSES + tid];
        out[g * OUT_CLASSES + tid] = o;
    }
}

// ---------------- launcher ----------------

extern "C" void kernel_launch(void* const* d_in, const int* in_sizes, int n_in,
                              void* d_out, int out_size, void* d_ws, size_t ws_size,
                              hipStream_t stream) {
    (void)in_sizes; (void)n_in; (void)out_size; (void)ws_size;

    const float* x     = (const float*)d_in[0];
    const int*   ei    = (const int*)  d_in[1];   // [2, E] flat
    const int*   batch = (const int*)  d_in[2];
    const float* W_in  = (const float*)d_in[3];
    const float* b_in  = (const float*)d_in[4];
    const float* W_hid = (const float*)d_in[5];   // [2,128,128]
    const float* b_hid = (const float*)d_in[6];   // [2,128]
    const float* Wm1   = (const float*)d_in[7];
    const float* bm1   = (const float*)d_in[8];
    const float* Wm2   = (const float*)d_in[9];
    const float* bm2   = (const float*)d_in[10];
    float* out = (float*)d_out;

    const int* src = ei;
    const int* dst = ei + N_EDGES;

    // workspace carve-up (~193 MB total)
    char* ws = (char*)d_ws;
    size_t off = 0;
    auto carve = [&](size_t bytes) -> char* {
        char* p = ws + off;
        off = (off + bytes + 255) & ~(size_t)255;
        return p;
    };
    float*          deg    = (float*)         carve((size_t)N_NODES * 4);
    float*          dinv   = (float*)         carve((size_t)N_NODES * 4);
    float*          invdeg = (float*)         carve((size_t)N_NODES * 4);
    float*          en     = (float*)         carve((size_t)N_EDGES * 4);
    unsigned short* hb     = (unsigned short*)carve((size_t)N_NODES * HID * 2);
    unsigned short* wpack  = (unsigned short*)carve((size_t)HID * HID * 2);
    float*          hW     = (float*)         carve((size_t)N_NODES * HID * 4);
    float*          agg    = (float*)         carve((size_t)N_NODES * HID * 4);
    float*          hC     = (float*)         carve((size_t)N_NODES * HID * 4);
    float*          sums   = (float*)         carve((size_t)NUM_GRAPHS * HID * 4);
    float*          cnt    = (float*)         carve((size_t)NUM_GRAPHS * 4);

    const int nodeBlocks  = (N_NODES + 255) / 256;
    const int edgeBlocks  = (N_EDGES + 255) / 256;
    const int feat4Count  = N_NODES * (HID / 4);
    const int feat4Blocks = (feat4Count + 255) / 256;
    const int edgeWaveBlocks = (int)(((long long)N_EDGES * 32 + 255) / 256); // 400000
    const int poolBlocks  = (int)(((long long)N_NODES * 32 + 255) / 256);

    // degrees + norms (reused by all 3 layers)
    k_init_deg <<<nodeBlocks, 256, 0, stream>>>(deg);
    k_deg_acc  <<<edgeBlocks, 256, 0, stream>>>(dst, deg);
    k_norms    <<<nodeBlocks, 256, 0, stream>>>(deg, dinv, invdeg);
    k_edge_norm<<<edgeBlocks, 256, 0, stream>>>(src, dst, dinv, en);

    // layer-0 input in bf16
    k_to_bf16_v4<<<feat4Blocks, 256, 0, stream>>>(x, hb, feat4Count);

    for (int l = 0; l < 3; ++l) {
        const float* W = (l == 0) ? W_in : (W_hid + (size_t)(l - 1) * HID * HID);
        const float* b = (l == 0) ? b_in : (b_hid + (size_t)(l - 1) * HID);

        k_pack_w      <<<(HID * HID + 255) / 256, 256, 0, stream>>>(W, wpack);
        k_gemm_wmma   <<<(N_TILES32 + 3) / 4, 128, 0, stream>>>(hb, wpack, invdeg,
                                                                hW, agg, N_TILES32);
        k_edge_scatter<<<edgeWaveBlocks, 256, 0, stream>>>(hW, en, src, dst, agg);
        k_relu_bias   <<<feat4Blocks, 256, 0, stream>>>(agg, b, hC, hb, (l < 2) ? 1 : 0);
    }

    // mean pool + head
    k_zero    <<<(NUM_GRAPHS * HID + 255) / 256, 256, 0, stream>>>(sums, NUM_GRAPHS * HID);
    k_zero    <<<1, 64, 0, stream>>>(cnt, NUM_GRAPHS);
    k_pool_acc<<<poolBlocks, 256, 0, stream>>>(hC, batch, sums);
    k_cnt_acc <<<nodeBlocks, 256, 0, stream>>>(batch, cnt);
    k_head    <<<NUM_GRAPHS, 128, 0, stream>>>(sums, cnt, Wm1, bm1, Wm2, bm2, out);
}
